// POINTCNN_SEG_attention_40475771797605
// MI455X (gfx1250) — compile-verified, matrix operand order checked by reading, not run on hardware
//
// PointCNN-seg forward for MI455X (gfx1250), wave32 / WMMA bf16.
// Dense GEMMs (l1b, lo, mlp_out*, fc1) run on v_wmma_f32_16x16x32_bf16 with f32
// accumulate; each wave computes a 16x64 output strip (4 N-tiles) so one A
// fragment feeds 4 WMMAs. KNN/FPS keep per-batch point sets in LDS. _mha is
// dead code in the reference and is skipped. Peak workspace ~550 MB from d_ws.

#include <hip/hip_runtime.h>

typedef __attribute__((ext_vector_type(16))) __bf16 bf16x16;
typedef __attribute__((ext_vector_type(8)))  __bf16 bf16x8;
typedef __attribute__((ext_vector_type(8)))  float  f32x8;

constexpr int BV  = 8;
constexpr int NP1 = 2048, NP2 = 768, NP3 = 288, NP4 = 108;
constexpr int NCLS = 13;

__device__ __forceinline__ float eluf(float x) { return x > 0.f ? x : expm1f(x); }
__device__ __forceinline__ float actf(float x, int a) {
  return a == 1 ? fmaxf(x, 0.f) : (a == 2 ? eluf(x) : x);
}

// ------------------------------------------------------------------ KNN ------
// One thread per query; database tile in LDS; KD stable min-scans (matches
// jax.lax.top_k(-d2) tie order); writes every dil-th neighbor (idx[..., ::dil]).
__global__ void knn_kernel(const float* __restrict__ qpos, const float* __restrict__ dpos,
                           int nq, int ndb, int KD, int dil, int Kout,
                           int* __restrict__ idxOut, float* __restrict__ distOut)
{
  extern __shared__ float sdb[];
  const int b = blockIdx.y;
  const float* dp = dpos + (size_t)b * ndb * 3;
  for (int j = threadIdx.x; j < ndb * 3; j += blockDim.x) sdb[j] = dp[j];
  __syncthreads();
  const int q = blockIdx.x * blockDim.x + threadIdx.x;
  if (q >= nq) return;
  const float* qp = qpos + ((size_t)b * nq + q) * 3;
  const float qx = qp[0], qy = qp[1], qz = qp[2];
  float lastD = -1.f; int lastI = -1;
  for (int s = 0; s < KD; ++s) {
    float bd = 3.4e38f; int bi = 0;
    for (int j = 0; j < ndb; ++j) {
      const float dx = sdb[3*j] - qx, dy = sdb[3*j+1] - qy, dz = sdb[3*j+2] - qz;
      const float d = dx*dx + dy*dy + dz*dz;
      const bool after = (d > lastD) || (d == lastD && j > lastI);
      if (after && d < bd) { bd = d; bi = j; }
    }
    lastD = bd; lastI = bi;
    if (s % dil == 0) {
      const size_t o = ((size_t)b * nq + q) * Kout + (size_t)(s / dil);
      idxOut[o] = bi;
      if (distOut) distOut[o] = bd;
    }
  }
}

// ------------------------------------------------------------------ FPS ------
__global__ void fps_kernel(const float* __restrict__ pos, int n, int m, int* __restrict__ idxOut)
{
  extern __shared__ float smem[];
  float* sp   = smem;            // n*3
  float* mind = sp + n * 3;      // n
  float* rv   = mind + n;        // blockDim
  int*   ri   = (int*)(rv + blockDim.x);
  __shared__ int lastS;
  const int b = blockIdx.x, tid = threadIdx.x;
  const float* gp = pos + (size_t)b * n * 3;
  for (int j = tid; j < n * 3; j += blockDim.x) sp[j] = gp[j];
  for (int j = tid; j < n; j += blockDim.x) mind[j] = 3.4e38f;
  if (tid == 0) { lastS = 0; idxOut[(size_t)b * m] = 0; }
  __syncthreads();
  for (int it = 1; it < m; ++it) {
    const int last = lastS;
    const float lx = sp[3*last], ly = sp[3*last+1], lz = sp[3*last+2];
    float bv = -1.f; int bi = 0;
    for (int j = tid; j < n; j += blockDim.x) {
      const float dx = sp[3*j]-lx, dy = sp[3*j+1]-ly, dz = sp[3*j+2]-lz;
      const float nm = fminf(mind[j], dx*dx + dy*dy + dz*dz);
      mind[j] = nm;
      if (nm > bv) { bv = nm; bi = j; }
    }
    rv[tid] = bv; ri[tid] = bi;
    __syncthreads();
    for (int o = blockDim.x >> 1; o > 0; o >>= 1) {
      if (tid < o) {
        if (rv[tid+o] > rv[tid] || (rv[tid+o] == rv[tid] && ri[tid+o] < ri[tid])) {
          rv[tid] = rv[tid+o]; ri[tid] = ri[tid+o];
        }
      }
      __syncthreads();
    }
    if (tid == 0) { lastS = ri[0]; idxOut[(size_t)b * m + it] = ri[0]; }
    __syncthreads();
  }
}

// ------------------------------------------------------------ elementwise ----
__global__ void gather_kernel(const float* __restrict__ src, const int* __restrict__ idx,
                              float* __restrict__ dst, int nin, int m, int F)
{
  const size_t t = (size_t)blockIdx.x * blockDim.x + threadIdx.x;
  const size_t total = (size_t)BV * m * F;
  if (t >= total) return;
  const int c = (int)(t % F);
  const size_t q = t / F;
  const int i = (int)(q % m), b = (int)(q / m);
  dst[t] = src[((size_t)b * nin + idx[(size_t)b * m + i]) * F + c];
}

__global__ void rel_kernel(const float* __restrict__ pos, const int* __restrict__ idx,
                           float* __restrict__ rel, int n, int K)
{
  const size_t t = (size_t)blockIdx.x * blockDim.x + threadIdx.x;
  const size_t total = (size_t)BV * n * K;
  if (t >= total) return;
  const size_t p = t / K;
  const int i = (int)(p % n), b = (int)(p / n);
  const int j = idx[t];
  const float* base = pos + (size_t)b * n * 3;
  rel[t*3+0] = base[3*j+0] - base[3*i+0];
  rel[t*3+1] = base[3*j+1] - base[3*i+1];
  rel[t*3+2] = base[3*j+2] - base[3*i+2];
}

__global__ void lin3_elu_kernel(const float* __restrict__ rel, const float* __restrict__ W,
                                const float* __restrict__ bias, float* __restrict__ out,
                                size_t rows, int cd)
{
  const size_t t = (size_t)blockIdx.x * blockDim.x + threadIdx.x;
  if (t >= rows * (size_t)cd) return;
  const size_t r = t / cd; const int c = (int)(t - r * cd);
  const float* rp = rel + r * 3;
  const float v = bias[c] + rp[0]*W[c] + rp[1]*W[cd + c] + rp[2]*W[2*cd + c];
  out[t] = eluf(v);
}

__global__ void l2_kernel(const float* __restrict__ rel2, const float* __restrict__ W,
                          const float* __restrict__ bias, float* __restrict__ out,
                          size_t P, int Kin, int K2)
{
  const size_t t = (size_t)blockIdx.x * blockDim.x + threadIdx.x;
  if (t >= P * (size_t)K2) return;
  const size_t p = t / K2; const int j = (int)(t - p * K2);
  float acc = bias[j];
  const float* rp = rel2 + p * Kin;
  for (int d = 0; d < Kin; ++d) acc += rp[d] * W[(size_t)d * K2 + j];
  out[t] = eluf(acc);
}

// t'[i,j] = act(sum_k t[i,k] * W[i,j,k] + b[i,j])
__global__ void c2_kernel(const float* __restrict__ tin, const float* __restrict__ W,
                          const float* __restrict__ bias, float* __restrict__ out,
                          size_t P, int K, int doElu)
{
  const int K2 = K * K;
  const size_t t = (size_t)blockIdx.x * blockDim.x + threadIdx.x;
  if (t >= P * (size_t)K2) return;
  const size_t p = t / K2; const int ij = (int)(t - p * K2);
  const int i = ij / K;
  float acc = bias[ij];
  const float* tp = tin + p * K2 + (size_t)i * K;
  const float* wp = W + (size_t)ij * K;
  for (int k = 0; k < K; ++k) acc += tp[k] * wp[k];
  out[t] = doElu ? eluf(acc) : acc;
}

// Fused xt = h^T t and d = sum_j xt[j,c] * dw[c,m,j]; h's x-part gathered on
// the fly (avoids materializing the (B,n,K,C) concat / xt tensors, ~400MB each).
__global__ void dmix_kernel(const float* __restrict__ hch, const float* __restrict__ xin,
                            const int* __restrict__ idx, const float* __restrict__ t3,
                            const float* __restrict__ dw, const float* __restrict__ db,
                            float* __restrict__ out, int n, int cin, int cd, int K, int dm)
{
  const int C = cin + cd;
  const size_t t = (size_t)blockIdx.x * blockDim.x + threadIdx.x;
  const size_t total = (size_t)BV * n * C;
  if (t >= total) return;
  const size_t p = t / C; const int c = (int)(t - p * C);
  const int b = (int)(p / n);
  const float* trow = t3 + p * (size_t)(K * K);
  for (int m = 0; m < dm; ++m) {
    float acc = db[(size_t)c * dm + m];
    const float* wp = dw + ((size_t)c * dm + m) * K;
    for (int k = 0; k < K; ++k) {
      float w = 0.f;
      const float* tk = trow + (size_t)k * K;
      for (int j = 0; j < K; ++j) w += tk[j] * wp[j];
      float hv;
      if (c < cd) hv = hch[((size_t)p * K + k) * cd + c];
      else        hv = xin[((size_t)b * n + idx[(size_t)p * K + k]) * cin + (c - cd)];
      acc += hv * w;
    }
    out[p * (size_t)(C * dm) + (size_t)c * dm + m] = acc;
  }
}

// ------------------------------------------------------------------- BN ------
__global__ void bn_reduce_kernel(const float* __restrict__ x, int ld, size_t rows,
                                 float* __restrict__ mean, float* __restrict__ istd)
{
  __shared__ float s1[256], s2[256];
  const int c = blockIdx.x, tid = threadIdx.x;
  float a = 0.f, q = 0.f;
  for (size_t r = tid; r < rows; r += 256) {
    const float v = x[r * (size_t)ld + c];
    a += v; q += v * v;
  }
  s1[tid] = a; s2[tid] = q;
  __syncthreads();
  for (int o = 128; o > 0; o >>= 1) {
    if (tid < o) { s1[tid] += s1[tid + o]; s2[tid] += s2[tid + o]; }
    __syncthreads();
  }
  if (tid == 0) {
    const float m = s1[0] / (float)rows;
    const float v = s2[0] / (float)rows - m * m;
    mean[c] = m; istd[c] = rsqrtf(v + 1e-5f);
  }
}

__global__ void bn_apply_kernel(float* __restrict__ x, int ld, size_t rows, int C,
                                const float* __restrict__ mean, const float* __restrict__ istd,
                                const float* __restrict__ g, const float* __restrict__ b)
{
  const size_t t = (size_t)blockIdx.x * blockDim.x + threadIdx.x;
  if (t >= rows * (size_t)C) return;
  const size_t r = t / C; const int c = (int)(t - r * C);
  const size_t o = r * (size_t)ld + c;
  x[o] = (x[o] - mean[c]) * istd[c] * g[c] + b[c];
}

// ------------------------------------------------------ bf16 conversions -----
__global__ void cvt_bf16_kernel(const float* __restrict__ src, int lda, __bf16* __restrict__ dst,
                                size_t rows, int K, int Kp)
{
  const size_t t = (size_t)blockIdx.x * blockDim.x + threadIdx.x;
  if (t >= rows * (size_t)Kp) return;
  const size_t r = t / Kp; const int k = (int)(t - r * Kp);
  dst[t] = (k < K) ? (__bf16)src[r * (size_t)lda + k] : (__bf16)0.f;
}

__global__ void cvt_add_bf16_kernel(const float* __restrict__ a, const float* __restrict__ b,
                                    __bf16* __restrict__ dst, size_t rows, int K, int Kp)
{
  const size_t t = (size_t)blockIdx.x * blockDim.x + threadIdx.x;
  if (t >= rows * (size_t)Kp) return;
  const size_t r = t / Kp; const int k = (int)(t - r * Kp);
  dst[t] = (k < K) ? (__bf16)(a[r * (size_t)K + k] + b[r * (size_t)K + k]) : (__bf16)0.f;
}

// W (K,N) row-major -> Wt (Np, Kp): B fragments then load contiguously along K.
__global__ void cvt_wt_kernel(const float* __restrict__ W, __bf16* __restrict__ dst,
                              int K, int N, int Kp, int Np)
{
  const size_t t = (size_t)blockIdx.x * blockDim.x + threadIdx.x;
  if (t >= (size_t)Np * Kp) return;
  const int n = (int)(t / Kp), k = (int)(t - (size_t)n * Kp);
  dst[t] = (n < N && k < K) ? (__bf16)W[(size_t)k * N + n] : (__bf16)0.f;
}

// ------------------------------------------------------------ WMMA GEMM ------
// One wave computes a 16x64 output strip = 4 N-tiles: one A fragment feeds 4
// v_wmma_f32_16x16x32_bf16 issues per 32-wide K step (2 A-loads + 8 B-loads ->
// 4 WMMAs), quartering A-side traffic vs one-tile-per-wave. ISA 16-bit A
// layout: lane (r=L%16, hi=L/16) holds row r, K chunks [hi*8,hi*8+8) and
// [16+hi*8,+8) -> two contiguous 16B loads per fragment. Wave-uniform early
// exit keeps EXEC all-ones through every WMMA issue.
__global__ void gemm_wmma_bf16(const __bf16* __restrict__ A, const __bf16* __restrict__ Bt,
                               const float* __restrict__ bias, float* __restrict__ C,
                               int M, int N, int Kp, int ldc, int act)
{
  const int lane = threadIdx.x & 31;
  const int wave = (int)(((size_t)blockIdx.x * blockDim.x + threadIdx.x) >> 5);
  const int tilesM = (M + 15) >> 4;
  const int tilesN = (N + 15) >> 4;
  const int tilesG = (tilesN + 3) >> 2;           // groups of 4 N-tiles
  if (wave >= tilesM * tilesG) return;            // wave-uniform
  const int tm = wave / tilesG;
  const int tg = wave - tm * tilesG;
  const int r  = lane & 15;
  const int hi = lane >> 4;
  int arow = tm * 16 + r; if (arow >= M) arow = M - 1;   // clamp; store is guarded
  const __bf16* ap = A + (size_t)arow * Kp + hi * 8;

  int tn0 = tg * 4 + 0; if (tn0 >= tilesN) tn0 = tilesN - 1;
  int tn1 = tg * 4 + 1; if (tn1 >= tilesN) tn1 = tilesN - 1;
  int tn2 = tg * 4 + 2; if (tn2 >= tilesN) tn2 = tilesN - 1;
  int tn3 = tg * 4 + 3; if (tn3 >= tilesN) tn3 = tilesN - 1;
  const __bf16* bp0 = Bt + (size_t)(tn0 * 16 + r) * Kp + hi * 8;
  const __bf16* bp1 = Bt + (size_t)(tn1 * 16 + r) * Kp + hi * 8;
  const __bf16* bp2 = Bt + (size_t)(tn2 * 16 + r) * Kp + hi * 8;
  const __bf16* bp3 = Bt + (size_t)(tn3 * 16 + r) * Kp + hi * 8;

  f32x8 acc0 = {0.f,0.f,0.f,0.f,0.f,0.f,0.f,0.f};
  f32x8 acc1 = acc0, acc2 = acc0, acc3 = acc0;

  for (int k = 0; k < Kp; k += 32) {
    bf16x8 a0 = *(const bf16x8*)(ap + k);
    bf16x8 a1 = *(const bf16x8*)(ap + k + 16);
    __builtin_prefetch(ap + k + 128, 0, 0);       // global_prefetch_b8
    bf16x16 av = __builtin_shufflevector(a0, a1, 0,1,2,3,4,5,6,7,8,9,10,11,12,13,14,15);

    bf16x8 q0 = *(const bf16x8*)(bp0 + k);
    bf16x8 q1 = *(const bf16x8*)(bp0 + k + 16);
    bf16x16 bv0 = __builtin_shufflevector(q0, q1, 0,1,2,3,4,5,6,7,8,9,10,11,12,13,14,15);
    acc0 = __builtin_amdgcn_wmma_f32_16x16x32_bf16(false, av, false, bv0, (short)0, acc0,
                                                   false, false);
    q0 = *(const bf16x8*)(bp1 + k);
    q1 = *(const bf16x8*)(bp1 + k + 16);
    bf16x16 bv1 = __builtin_shufflevector(q0, q1, 0,1,2,3,4,5,6,7,8,9,10,11,12,13,14,15);
    acc1 = __builtin_amdgcn_wmma_f32_16x16x32_bf16(false, av, false, bv1, (short)0, acc1,
                                                   false, false);
    q0 = *(const bf16x8*)(bp2 + k);
    q1 = *(const bf16x8*)(bp2 + k + 16);
    bf16x16 bv2 = __builtin_shufflevector(q0, q1, 0,1,2,3,4,5,6,7,8,9,10,11,12,13,14,15);
    acc2 = __builtin_amdgcn_wmma_f32_16x16x32_bf16(false, av, false, bv2, (short)0, acc2,
                                                   false, false);
    q0 = *(const bf16x8*)(bp3 + k);
    q1 = *(const bf16x8*)(bp3 + k + 16);
    bf16x16 bv3 = __builtin_shufflevector(q0, q1, 0,1,2,3,4,5,6,7,8,9,10,11,12,13,14,15);
    acc3 = __builtin_amdgcn_wmma_f32_16x16x32_bf16(false, av, false, bv3, (short)0, acc3,
                                                   false, false);
  }

  // Epilogue: D layout m = v + 8*hi, n = lane%16. Lane-divergent guards are
  // fine here (no further WMMA issues).
  const int mbase = tm * 16 + 8 * hi;
  {
    const int nc = tg * 64 + 0 * 16 + r;
    if (nc < N) {
      const float bb = bias ? bias[nc] : 0.f;
      #pragma unroll
      for (int v = 0; v < 8; ++v) {
        const int mrow = mbase + v;
        if (mrow < M) C[(size_t)mrow * ldc + nc] = actf(acc0[v] + bb, act);
      }
    }
  }
  {
    const int nc = tg * 64 + 1 * 16 + r;
    if (nc < N) {
      const float bb = bias ? bias[nc] : 0.f;
      #pragma unroll
      for (int v = 0; v < 8; ++v) {
        const int mrow = mbase + v;
        if (mrow < M) C[(size_t)mrow * ldc + nc] = actf(acc1[v] + bb, act);
      }
    }
  }
  {
    const int nc = tg * 64 + 2 * 16 + r;
    if (nc < N) {
      const float bb = bias ? bias[nc] : 0.f;
      #pragma unroll
      for (int v = 0; v < 8; ++v) {
        const int mrow = mbase + v;
        if (mrow < M) C[(size_t)mrow * ldc + nc] = actf(acc2[v] + bb, act);
      }
    }
  }
  {
    const int nc = tg * 64 + 3 * 16 + r;
    if (nc < N) {
      const float bb = bias ? bias[nc] : 0.f;
      #pragma unroll
      for (int v = 0; v < 8; ++v) {
        const int mrow = mbase + v;
        if (mrow < M) C[(size_t)mrow * ldc + nc] = actf(acc3[v] + bb, act);
      }
    }
  }
}

// ------------------------------------------------------------- interp/fc -----
__global__ void interp_kernel(const float* __restrict__ x, const int* __restrict__ idx3,
                              const float* __restrict__ d3, float* __restrict__ out,
                              int nx, int ny, int F)
{
  const size_t t = (size_t)blockIdx.x * blockDim.x + threadIdx.x;
  const size_t total = (size_t)BV * ny * F;
  if (t >= total) return;
  const int c = (int)(t % F);
  const size_t q = t / F;
  const int i = (int)(q % ny), b = (int)(q / ny);
  const int*   ip = idx3 + ((size_t)b * ny + i) * 3;
  const float* dp = d3   + ((size_t)b * ny + i) * 3;
  float ws = 0.f, acc = 0.f;
  for (int j = 0; j < 3; ++j) {
    const float w = 1.f / fmaxf(dp[j], 1e-16f);
    ws += w;
    acc += w * x[((size_t)b * nx + ip[j]) * F + c];
  }
  out[t] = acc / ws;
}

__global__ void fc2_kernel(const float* __restrict__ fcm, const float* __restrict__ w,
                           const float* __restrict__ bias, float* __restrict__ out,
                           int n, int Kin, int ncls)
{
  const size_t t = (size_t)blockIdx.x * blockDim.x + threadIdx.x;
  const size_t total = (size_t)BV * n * ncls;
  if (t >= total) return;
  const int cls = (int)(t % ncls);
  const size_t q = t / ncls;
  const int i = (int)(q % n), b = (int)(q / n);
  float acc = bias[cls];
  const float* fp = fcm + q * Kin;
  for (int k = 0; k < Kin; ++k) acc += fp[k] * w[(size_t)k * ncls + cls];
  out[((size_t)b * ncls + cls) * n + i] = acc;   // transposed (B, 13, N) store
}

// =============================== host side ===================================
static inline int pad16(int x) { return (x + 15) & ~15; }
static inline int pad32(int x) { return (x + 31) & ~31; }
static inline int nblk(size_t n) { return (int)((n + 255) / 256); }

struct WS {
  float *HCH, *REL, *T0, *T1, *DBUF, *MEAN, *ISTD, *DIST3;
  __bf16 *ABF, *WT;
  int *IDX, *IDX3, *FPSI;
};

// Sorted sub-keys of each conv dict (JAX pytree flattening order).
enum { BN1A_B, BN1A_G, BN1B_B, BN1B_G, BN2A_B, BN2A_G, BN2B_B, BN2B_G, BN2C_B, BN2C_G,
       C2A_B, C2A_W, C2B_B, C2B_W, DB_, DW_, L1A_B, L1A_W, L1B_B, L1B_W, L2_B, L2_W,
       LO_B, LO_W };
struct ConvP { const float* p[24]; };

static void bn(float* x, size_t rows, int C, int ld, const float* g, const float* b,
               WS& ws, hipStream_t s)
{
  bn_reduce_kernel<<<C, 256, 0, s>>>(x, ld, rows, ws.MEAN, ws.ISTD);
  bn_apply_kernel<<<nblk(rows * (size_t)C), 256, 0, s>>>(x, ld, rows, C, ws.MEAN, ws.ISTD, g, b);
}

static void gemm_launch(const __bf16* A, const __bf16* Bt, const float* bias, float* C,
                        int M, int N, int Kp, int ldc, int act, hipStream_t s)
{
  const int tilesM = (M + 15) >> 4;
  const int tilesG = (((N + 15) >> 4) + 3) >> 2;
  const size_t waves = (size_t)tilesM * tilesG;
  gemm_wmma_bf16<<<nblk(waves * 32), 256, 0, s>>>(A, Bt, bias, C, M, N, Kp, ldc, act);
}

static void gemm_f32(const float* A, int lda, const float* W, const float* bias,
                     float* C, int ldc, int M, int N, int K, int act, WS& ws, hipStream_t s)
{
  const int Kp = pad32(K), Np = pad16(N);
  cvt_bf16_kernel<<<nblk((size_t)M * Kp), 256, 0, s>>>(A, lda, ws.ABF, (size_t)M, K, Kp);
  cvt_wt_kernel<<<nblk((size_t)Np * Kp), 256, 0, s>>>(W, ws.WT, K, N, Kp, Np);
  gemm_launch(ws.ABF, ws.WT, bias, C, M, N, Kp, ldc, act, s);
}

static void gemm_add_f32(const float* A0, const float* A1, const float* W, const float* bias,
                         float* C, int M, int N, int K, int act, WS& ws, hipStream_t s)
{
  const int Kp = pad32(K), Np = pad16(N);
  cvt_add_bf16_kernel<<<nblk((size_t)M * Kp), 256, 0, s>>>(A0, A1, ws.ABF, (size_t)M, K, Kp);
  cvt_wt_kernel<<<nblk((size_t)Np * Kp), 256, 0, s>>>(W, ws.WT, K, N, Kp, Np);
  gemm_launch(ws.ABF, ws.WT, bias, C, M, N, Kp, N, act, s);
}

static void run_xconv(const ConvP& P, const float* xin, const float* pos, float* out,
                      int n, int cin, int cd, int cout, int K, int dil, WS& ws, hipStream_t s)
{
  const int C = cin + cd;
  const int dm = (cout + C - 1) / C;
  const size_t Pn = (size_t)BV * n;
  const size_t rows = Pn * K;
  const int K2 = K * K;

  dim3 kg((unsigned)((n + 127) / 128), (unsigned)BV);
  knn_kernel<<<kg, 128, (size_t)n * 3 * sizeof(float), s>>>(pos, pos, n, n, K * dil, dil, K,
                                                            ws.IDX, nullptr);
  rel_kernel<<<nblk(rows), 256, 0, s>>>(pos, ws.IDX, ws.REL, n, K);

  // h chain: lin3 -> elu -> BN -> (WMMA) cd x cd GEMM -> elu -> BN
  lin3_elu_kernel<<<nblk(rows * (size_t)cd), 256, 0, s>>>(ws.REL, P.p[L1A_W], P.p[L1A_B],
                                                          ws.HCH, rows, cd);
  bn(ws.HCH, rows, cd, cd, P.p[BN1A_G], P.p[BN1A_B], ws, s);
  gemm_f32(ws.HCH, cd, P.p[L1B_W], P.p[L1B_B], ws.HCH, cd, (int)rows, cd, cd, 2 /*elu*/, ws, s);
  bn(ws.HCH, rows, cd, cd, P.p[BN1B_G], P.p[BN1B_B], ws, s);

  // t chain: rel(B*n, 3K) @ l2 -> BN -> c2a -> BN -> c2b -> BN
  l2_kernel<<<nblk(Pn * (size_t)K2), 256, 0, s>>>(ws.REL, P.p[L2_W], P.p[L2_B], ws.T0,
                                                  Pn, 3 * K, K2);
  bn(ws.T0, Pn, K2, K2, P.p[BN2A_G], P.p[BN2A_B], ws, s);
  c2_kernel<<<nblk(Pn * (size_t)K2), 256, 0, s>>>(ws.T0, P.p[C2A_W], P.p[C2A_B], ws.T1, Pn, K, 1);
  bn(ws.T1, Pn, K2, K2, P.p[BN2B_G], P.p[BN2B_B], ws, s);
  c2_kernel<<<nblk(Pn * (size_t)K2), 256, 0, s>>>(ws.T1, P.p[C2B_W], P.p[C2B_B], ws.T0, Pn, K, 0);
  bn(ws.T0, Pn, K2, K2, P.p[BN2C_G], P.p[BN2C_B], ws, s);

  // fused xt + depthwise mix, then lo GEMM with fused ReLU (relu(_xconv) in ref)
  dmix_kernel<<<nblk(Pn * (size_t)C), 256, 0, s>>>(ws.HCH, xin, ws.IDX, ws.T0,
                                                   P.p[DW_], P.p[DB_], ws.DBUF,
                                                   n, cin, cd, K, dm);
  gemm_f32(ws.DBUF, C * dm, P.p[LO_W], P.p[LO_B], out, cout, (int)Pn, cout, C * dm,
           1 /*relu*/, ws, s);
}

static void downsample(const float* posIn, const float* xIn, float* posOut, float* xOut,
                       int nIn, int m, int F, WS& ws, hipStream_t s)
{
  const size_t lds = (size_t)nIn * 16 + 2048;
  fps_kernel<<<BV, 256, lds, s>>>(posIn, nIn, m, ws.FPSI);
  gather_kernel<<<nblk((size_t)BV * m * 3), 256, 0, s>>>(posIn, ws.FPSI, posOut, nIn, m, 3);
  gather_kernel<<<nblk((size_t)BV * m * F), 256, 0, s>>>(xIn, ws.FPSI, xOut, nIn, m, F);
}

static void knn_interp(const float* x, const float* posx, const float* posy, float* out,
                       int nx, int ny, int F, WS& ws, hipStream_t s)
{
  dim3 g((unsigned)((ny + 127) / 128), (unsigned)BV);
  knn_kernel<<<g, 128, (size_t)nx * 3 * sizeof(float), s>>>(posy, posx, ny, nx, 3, 1, 3,
                                                            ws.IDX3, ws.DIST3);
  interp_kernel<<<nblk((size_t)BV * ny * F), 256, 0, s>>>(x, ws.IDX3, ws.DIST3, out, nx, ny, F);
}

extern "C" void kernel_launch(void* const* d_in, const int* in_sizes, int n_in,
                              void* d_out, int out_size, void* d_ws, size_t ws_size,
                              hipStream_t stream)
{
  (void)in_sizes; (void)n_in; (void)out_size; (void)ws_size;

  // Input map: JAX pytree flattening (dict keys sorted alphabetically, DFS).
  // 0: points; 1-4: attn {in_b,in_w,out_b,out_w} (dead); 5..: conv1..conv4,
  // conv_up1..conv_up4 (24 tensors each); 197 fc1_w; 198 fc2_b; 199 fc2_w;
  // 200..207: mlp_out1..mlp_out4 {b,w}.
  const float* pos1 = (const float*)d_in[0];
  ConvP c1, c2, c3, c4, u1, u2, u3, u4;
  for (int j = 0; j < 24; ++j) {
    c1.p[j] = (const float*)d_in[5   + j];
    c2.p[j] = (const float*)d_in[29  + j];
    c3.p[j] = (const float*)d_in[53  + j];
    c4.p[j] = (const float*)d_in[77  + j];
    u1.p[j] = (const float*)d_in[101 + j];
    u2.p[j] = (const float*)d_in[125 + j];
    u3.p[j] = (const float*)d_in[149 + j];
    u4.p[j] = (const float*)d_in[173 + j];
  }
  const float* fc1_w = (const float*)d_in[197];
  const float* fc2_b = (const float*)d_in[198];
  const float* fc2_w = (const float*)d_in[199];
  const float* m1_b = (const float*)d_in[200]; const float* m1_w = (const float*)d_in[201];
  const float* m2_b = (const float*)d_in[202]; const float* m2_w = (const float*)d_in[203];
  const float* m3_b = (const float*)d_in[204]; const float* m3_w = (const float*)d_in[205];
  const float* m4_b = (const float*)d_in[206]; const float* m4_w = (const float*)d_in[207];

  // -------- workspace carve (bump allocator, 256B aligned) --------
  char* base = (char*)d_ws;
  size_t off = 0;
  auto takeB = [&](size_t bytes) -> void* {
    void* p = base + off;
    off = (off + bytes + 255) & ~((size_t)255);
    return p;
  };
  auto takeF = [&](size_t elems) -> float*  { return (float*)takeB(elems * 4); };
  auto takeH = [&](size_t elems) -> __bf16* { return (__bf16*)takeB(elems * 2); };
  auto takeI = [&](size_t elems) -> int*    { return (int*)takeB(elems * 4); };

  WS ws;
  ws.HCH  = takeF(48ull << 20);   // max rows*cd = 73728*640 = 47.2M f32
  ws.ABF  = takeH(48ull << 20);   // bf16 A snapshot, same cap
  ws.WT   = takeH(2ull  << 20);   // max Np*Kp = 1024*1920
  ws.DBUF = takeF(11ull << 20);   // max B*n*C*dm = 16384*632
  ws.T0   = takeF(1ull  << 20);   // max B*n*K^2 = 16384*64
  ws.T1   = takeF(1ull  << 20);
  ws.REL  = takeF((size_t)BV * NP1 * 8 * 3);
  ws.IDX  = takeI((size_t)BV * NP1 * 16);
  ws.IDX3 = takeI((size_t)BV * NP1 * 3);
  ws.DIST3= takeF((size_t)BV * NP1 * 3);
  ws.FPSI = takeI((size_t)BV * NP2);
  ws.MEAN = takeF(1024);
  ws.ISTD = takeF(1024);

  float* x1   = takeF((size_t)BV * NP1 * 256);
  float* pos2 = takeF((size_t)BV * NP2 * 3);
  float* x2in = takeF((size_t)BV * NP2 * 256);
  float* x2   = takeF((size_t)BV * NP2 * 512);
  float* pos3 = takeF((size_t)BV * NP3 * 3);
  float* x3in = takeF((size_t)BV * NP3 * 512);
  float* x3   = takeF((size_t)BV * NP3 * 768);
  float* pos4 = takeF((size_t)BV * NP4 * 3);
  float* x4in = takeF((size_t)BV * NP4 * 768);
  float* x4   = takeF((size_t)BV * NP4 * 1024);
  float* xo4  = takeF((size_t)BV * NP4 * 1024);
  float* h4   = takeF((size_t)BV * NP4 * 1024);
  float* xi3  = takeF((size_t)BV * NP3 * 1024);
  float* xo3  = takeF((size_t)BV * NP3 * 768);
  float* h3   = takeF((size_t)BV * NP3 * 768);
  float* xi2  = takeF((size_t)BV * NP2 * 768);
  float* xo2  = takeF((size_t)BV * NP2 * 512);
  float* h2   = takeF((size_t)BV * NP2 * 512);
  float* xi1  = takeF((size_t)BV * NP1 * 512);
  float* xo1  = takeF((size_t)BV * NP1 * 256);
  float* h1   = takeF((size_t)BV * NP1 * 256);
  float* fcm  = takeF((size_t)BV * NP1 * 128);

  // ------------------------------ encoder ------------------------------
  run_xconv(c1, nullptr, pos1, x1, NP1,    0, 128,  256,  8, 1, ws, stream);
  downsample(pos1, x1, pos2, x2in, NP1, NP2, 256, ws, stream);
  run_xconv(c2, x2in, pos2, x2, NP2,  256, 384,  512, 12, 2, ws, stream);
  downsample(pos2, x2, pos3, x3in, NP2, NP3, 512, ws, stream);
  run_xconv(c3, x3in, pos3, x3, NP3,  512, 640,  768, 16, 2, ws, stream);
  downsample(pos3, x3, pos4, x4in, NP3, NP4, 768, ws, stream);
  run_xconv(c4, x4in, pos4, x4, NP4,  768, 896, 1024, 16, 4, ws, stream);

  // _mha(x4) result is dead in the reference -> skipped.

  // ------------------------------ decoder ------------------------------
  run_xconv(u4, x4, pos4, xo4, NP4, 1024, 896, 1024, 16, 4, ws, stream);
  gemm_add_f32(xo4, x4, m4_w, m4_b, h4, BV * NP4, 1024, 1024, 0, ws, stream);
  knn_interp(h4, pos4, pos3, xi3, NP4, NP3, 1024, ws, stream);

  run_xconv(u3, xi3, pos3, xo3, NP3, 1024, 896, 768, 16, 2, ws, stream);
  gemm_add_f32(xo3, x3, m3_w, m3_b, h3, BV * NP3, 768, 768, 0, ws, stream);
  knn_interp(h3, pos3, pos2, xi2, NP3, NP2, 768, ws, stream);

  run_xconv(u2, xi2, pos2, xo2, NP2, 768, 640, 512, 12, 2, ws, stream);
  gemm_add_f32(xo2, x2, m2_w, m2_b, h2, BV * NP2, 512, 512, 0, ws, stream);
  knn_interp(h2, pos2, pos1, xi1, NP2, NP1, 512, ws, stream);

  run_xconv(u1, xi1, pos1, xo1, NP1, 512, 120, 256, 8, 2, ws, stream);
  gemm_add_f32(xo1, x1, m1_w, m1_b, h1, BV * NP1, 256, 256, 0, ws, stream);

  // head: relu(h1 @ fc1_w) @ fc2_w + fc2_b, transposed store (B, 13, N)
  gemm_f32(h1, 256, fc1_w, nullptr, fcm, 128, BV * NP1, 128, 256, 1 /*relu*/, ws, stream);
  fc2_kernel<<<nblk((size_t)BV * NP1 * NCLS), 256, 0, stream>>>(fcm, fc2_w, fc2_b,
                                                                (float*)d_out, NP1, 128, NCLS);
}
